// Uncond_MambaPolicy_72928544686654
// MI455X (gfx1250) — compile-verified, matrix operand order checked by reading, not run on hardware
//
#include <hip/hip_runtime.h>
#include <hip/hip_bf16.h>
#include <math.h>

// ---------------------------------------------------------------------------
// Mamba policy forward for gfx1250 (MI455X).
//  * All activation GEMMs: v_wmma_f32_16x16x32_f16, 16x64 strip per wave
//    (4 accumulators share one A fragment -> 4x less A traffic, 4 WMMAs
//    back-to-back per K-step).
//  * dt projection (K=16): v_wmma_f32_16x16x4_f32 (full fp32 matrix path).
//  * Selective scan: register-resident VALU scan; B/C chunks staged into LDS
//    via global_load_async_to_lds_b32 (ASYNCcnt) instead of VGPR round-trips.
// ---------------------------------------------------------------------------

typedef __attribute__((ext_vector_type(16))) _Float16 v16h;
typedef __attribute__((ext_vector_type(8)))  float    v8f;
typedef __attribute__((ext_vector_type(2)))  float    v2f;

#define BATCH   4
#define SEQ     1024
#define DIN     64
#define DMODEL  256
#define DINNER  512
#define DTRANK  16
#define NSTATE  16
#define NLAYER  4
#define DCONV   4
#define MROWS   (BATCH * SEQ)      // 4096 token rows

enum { EPI_STORE = 0, EPI_BIAS = 1, EPI_BIAS_RELU = 2, EPI_ADD = 3, EPI_EMB = 4 };

// ---------------------------------------------------------------------------
// WMMA GEMM: Out[M x N] = A[M x K] * W[N x K]^T (+ epilogue)
// One wave computes a 16 x (16*NT) strip; 8 waves / 256-thread block.
// A-frag: lane r=lane&15 holds row tm+r; K halves selected by lane>>4 per the
// ISA 16-bit A layout.  B-frag: lane holds column tn+(lane&15) of B == a row
// of W (contiguous).  C/D: acc[i] -> m = tm + 8*(lane>>4) + i, n = lane&15.
// ---------------------------------------------------------------------------
template <int EPI, int NT>
__global__ void gemm_wmma_kernel(const float* __restrict__ A,
                                 const float* __restrict__ W,
                                 const float* __restrict__ bias,
                                 const float* __restrict__ Cin,
                                 float* __restrict__ Out,
                                 int Ndim, int K)
{
    const int wave  = threadIdx.x >> 5;
    const int lane  = threadIdx.x & 31;
    const int group = blockIdx.x * 8 + wave;
    const int ngn   = Ndim / (16 * NT);
    const int tm    = (group / ngn) << 4;
    const int tn0   = (group % ngn) * (16 * NT);
    const int r     = lane & 15;
    const int kb    = (lane >> 4) << 3;   // 0 or 8

    const float* Arow = A + (tm + r) * K;

    v8f acc[NT];
#pragma unroll
    for (int t = 0; t < NT; ++t) acc[t] = (v8f){};

    for (int k0 = 0; k0 < K; k0 += 32) {
        v16h af;
#pragma unroll
        for (int j = 0; j < 8; ++j) {
            af[j]     = (_Float16)Arow[k0 + kb + j];
            af[j + 8] = (_Float16)Arow[k0 + kb + 16 + j];
        }
#pragma unroll
        for (int t = 0; t < NT; ++t) {
            const float* Wrow = W + (tn0 + 16 * t + r) * K;
            v16h bf;
#pragma unroll
            for (int j = 0; j < 8; ++j) {
                bf[j]     = (_Float16)Wrow[k0 + kb + j];
                bf[j + 8] = (_Float16)Wrow[k0 + kb + 16 + j];
            }
            acc[t] = __builtin_amdgcn_wmma_f32_16x16x32_f16(
                false, af, false, bf, (short)0, acc[t], false, false);
        }
    }

    const int mo = (lane >> 4) << 3;
#pragma unroll
    for (int t = 0; t < NT; ++t) {
        const int n = tn0 + 16 * t + (lane & 15);
#pragma unroll
        for (int i = 0; i < 8; ++i) {
            int   m = tm + mo + i;
            float v = acc[t][i];
            if (EPI == EPI_BIAS || EPI == EPI_BIAS_RELU || EPI == EPI_EMB) v += bias[n];
            if (EPI == EPI_BIAS_RELU) v = fmaxf(v, 0.0f);
            if (EPI == EPI_ADD) v += Cin[m * Ndim + n];
            if (EPI == EPI_EMB) {   // sinusoidal positional encoding, row m -> t
                int   tt  = m & (SEQ - 1);
                float dv  = __expf(-(float)(n & ~1) * (9.210340371976184f / (float)DMODEL));
                float ang = (float)tt * dv;
                v += (n & 1) ? __cosf(ang) : __sinf(ang);
            }
            Out[m * Ndim + n] = v;
        }
    }
}

// ---------------------------------------------------------------------------
// dt = softplus(dbc[:, :16] @ dt_w^T + dt_b)   via V_WMMA_F32_16X16X4_F32
// A = dbc rows (stride 48, first 16 cols), W = dt_w (DINNER x 16).
// fp32 A layout: lane r holds row, VGPR0/1 = K=(kb,kb+1) with kb=(lane>>4)*2.
// ---------------------------------------------------------------------------
__global__ void dt_wmma_kernel(const float* __restrict__ dbc,  // MROWS x 48
                               const float* __restrict__ dtw,  // DINNER x 16
                               const float* __restrict__ dtb,  // DINNER
                               float* __restrict__ dt)         // MROWS x DINNER
{
    const int wave  = threadIdx.x >> 5;
    const int lane  = threadIdx.x & 31;
    const int group = blockIdx.x * 8 + wave;
    const int ngn   = DINNER / 64;        // 8 groups of 4 n-tiles
    const int tm    = (group / ngn) << 4;
    const int tn0   = (group % ngn) << 6;
    const int r     = lane & 15;
    const int kb    = (lane >> 4) << 1;   // 0 or 2

    const float* Arow = dbc + (tm + r) * 48;

    v8f acc[4];
#pragma unroll
    for (int t = 0; t < 4; ++t) acc[t] = (v8f){};

#pragma unroll
    for (int k0 = 0; k0 < DTRANK; k0 += 4) {
        v2f a;
        a[0] = Arow[k0 + kb];
        a[1] = Arow[k0 + kb + 1];
#pragma unroll
        for (int t = 0; t < 4; ++t) {
            const float* Wrow = dtw + (tn0 + 16 * t + r) * DTRANK;
            v2f bv;
            bv[0] = Wrow[k0 + kb];
            bv[1] = Wrow[k0 + kb + 1];
            acc[t] = __builtin_amdgcn_wmma_f32_16x16x4_f32(
                false, a, false, bv, (short)0, acc[t], false, false);
        }
    }

    const int mo = (lane >> 4) << 3;
#pragma unroll
    for (int t = 0; t < 4; ++t) {
        const int n = tn0 + 16 * t + (lane & 15);
#pragma unroll
        for (int i = 0; i < 8; ++i) {
            int   m = tm + mo + i;
            float v = acc[t][i] + dtb[n];
            dt[m * DINNER + n] = (v > 20.0f) ? v : __logf(1.0f + __expf(v));   // softplus
        }
    }
}

// --------------------------- LayerNorm (in-place) ---------------------------
__global__ void layernorm_kernel(float* __restrict__ h,
                                 const float* __restrict__ w,
                                 const float* __restrict__ b)
{
    __shared__ float s[DMODEL];
    const int row = blockIdx.x, tid = threadIdx.x;
    float v = h[row * DMODEL + tid];
    s[tid] = v; __syncthreads();
    for (int st = DMODEL / 2; st > 0; st >>= 1) {
        if (tid < st) s[tid] += s[tid + st];
        __syncthreads();
    }
    float mu = s[0] * (1.0f / DMODEL);
    __syncthreads();
    float d = v - mu;
    s[tid] = d * d; __syncthreads();
    for (int st = DMODEL / 2; st > 0; st >>= 1) {
        if (tid < st) s[tid] += s[tid + st];
        __syncthreads();
    }
    float var = s[0] * (1.0f / DMODEL);
    h[row * DMODEL + tid] = d * rsqrtf(var + 1e-5f) * w[tid] + b[tid];
}

// ------------------------------- RMSNorm ------------------------------------
__global__ void rmsnorm_kernel(const float* __restrict__ h,
                               const float* __restrict__ w,
                               float* __restrict__ out)
{
    __shared__ float s[DMODEL];
    const int row = blockIdx.x, tid = threadIdx.x;
    float v = h[row * DMODEL + tid];
    s[tid] = v * v; __syncthreads();
    for (int st = DMODEL / 2; st > 0; st >>= 1) {
        if (tid < st) s[tid] += s[tid + st];
        __syncthreads();
    }
    out[row * DMODEL + tid] = v * rsqrtf(s[0] * (1.0f / DMODEL) + 1e-5f) * w[tid];
}

// ------------------ depthwise causal conv (k=4) + bias + SiLU ---------------
__global__ void conv_silu_kernel(const float* __restrict__ xz,   // [B,L,2*DINNER]
                                 const float* __restrict__ cw,   // DINNER x 4
                                 const float* __restrict__ cb,   // DINNER
                                 float* __restrict__ xc)         // [B,L,DINNER]
{
    const int idx = blockIdx.x * blockDim.x + threadIdx.x;   // over MROWS*DINNER
    const int d = idx & (DINNER - 1);
    const int m = idx >> 9;
    const int t = m & (SEQ - 1);
    const int b = m >> 10;
    float acc = cb[d];
#pragma unroll
    for (int j = 0; j < DCONV; ++j) {
        int tt = t + j - (DCONV - 1);
        if (tt >= 0)
            acc += cw[d * DCONV + j] * xz[(((b << 10) + tt) * (2 * DINNER)) + d];
    }
    xc[idx] = acc * (1.0f / (1.0f + __expf(-acc)));          // SiLU
}

// ------------------------------ selective scan ------------------------------
// grid: (DINNER/256, BATCH); one channel d per thread, 16 fp32 states in VGPRs.
// B/C chunks (16 timesteps) staged into LDS through the gfx1250 async path:
// global_load_async_to_lds_b32 + s_wait_asynccnt + block barrier.
__global__ void scan_kernel(const float* __restrict__ dt,    // MROWS x DINNER
                            const float* __restrict__ xc,    // MROWS x DINNER
                            const float* __restrict__ dbc,   // MROWS x 48 (B@16, C@32)
                            const float* __restrict__ Alog,  // DINNER x 16
                            float* __restrict__ ys)          // MROWS x DINNER
{
    const int b = blockIdx.y;
    const int d = blockIdx.x * 256 + threadIdx.x;
    float A[NSTATE], h[NSTATE];
#pragma unroll
    for (int n = 0; n < NSTATE; ++n) {
        A[n] = -__expf(Alog[d * NSTATE + n]);
        h[n] = 0.0f;
    }
    __shared__ float Bs[16][NSTATE];
    __shared__ float Cs[16][NSTATE];
    const int tt = threadIdx.x >> 4;
    const int nn = threadIdx.x & 15;
    for (int t0 = 0; t0 < SEQ; t0 += 16) {
        __syncthreads();
        const int rbase = ((b << 10) + t0 + tt) * 48;
        // async global -> LDS (bypasses VGPRs, tracked with ASYNCcnt)
        asm volatile("global_load_async_to_lds_b32 %0, %1, off"
                     :: "v"((unsigned)(unsigned long long)&Bs[tt][nn]),
                        "v"(dbc + rbase + 16 + nn)
                     : "memory");
        asm volatile("global_load_async_to_lds_b32 %0, %1, off"
                     :: "v"((unsigned)(unsigned long long)&Cs[tt][nn]),
                        "v"(dbc + rbase + 32 + nn)
                     : "memory");
        asm volatile("s_wait_asynccnt 0x0" ::: "memory");
        __syncthreads();
        for (int t = 0; t < 16; ++t) {
            const int   m   = (b << 10) + t0 + t;
            const float dtv = dt[m * DINNER + d];
            const float xv  = xc[m * DINNER + d];
            float y = 0.0f;
#pragma unroll
            for (int n = 0; n < NSTATE; ++n) {
                float dA = __expf(dtv * A[n]);
                h[n] = dA * h[n] + dtv * Bs[t][n] * xv;
                y += h[n] * Cs[t][n];
            }
            ys[m * DINNER + d] = y;
        }
    }
}

// --------------------- y = (ys + Dp*xc) * silu(z) ---------------------------
__global__ void combine_kernel(const float* __restrict__ ys,
                               const float* __restrict__ xc,
                               const float* __restrict__ xz,   // z = cols [512,1024)
                               const float* __restrict__ Dp,
                               float* __restrict__ y)
{
    const int idx = blockIdx.x * blockDim.x + threadIdx.x;
    const int d = idx & (DINNER - 1);
    const int m = idx >> 9;
    const float z = xz[m * (2 * DINNER) + DINNER + d];
    y[idx] = (ys[idx] + Dp[d] * xc[idx]) * (z * (1.0f / (1.0f + __expf(-z))));
}

// ---------------------------------------------------------------------------
extern "C" void kernel_launch(void* const* d_in, const int* in_sizes, int n_in,
                              void* d_out, int out_size, void* d_ws, size_t ws_size,
                              hipStream_t stream)
{
    (void)in_sizes; (void)n_in; (void)out_size; (void)ws_size;

    const float* x        = (const float*)d_in[0];
    const float* emb_w    = (const float*)d_in[1];
    const float* emb_b    = (const float*)d_in[2];
    const float* ln_w     = (const float*)d_in[3];
    const float* ln_b     = (const float*)d_in[4];
    const float* in_proj  = (const float*)d_in[5];   // NL x 1024 x 256
    const float* conv_w   = (const float*)d_in[6];   // NL x 512 x 4
    const float* conv_b   = (const float*)d_in[7];   // NL x 512
    const float* x_proj   = (const float*)d_in[8];   // NL x 48 x 512
    const float* dt_w     = (const float*)d_in[9];   // NL x 512 x 16
    const float* dt_b     = (const float*)d_in[10];  // NL x 512
    const float* A_log    = (const float*)d_in[11];  // NL x 512 x 16
    const float* Dp       = (const float*)d_in[12];  // NL x 512
    const float* out_proj = (const float*)d_in[13];  // NL x 256 x 512
    const float* norm_w   = (const float*)d_in[14];  // NL x 256
    const float* normf_w  = (const float*)d_in[15];
    const float* h1_w     = (const float*)d_in[16];  // 128 x 256
    const float* h1_b     = (const float*)d_in[17];
    const float* h2_w     = (const float*)d_in[18];  // 64 x 128
    const float* h2_b     = (const float*)d_in[19];
    float* outp = (float*)d_out;

    // workspace layout (~62 MB of fp32)
    float* res  = (float*)d_ws;                      // 4096 x 256
    float* hn   = res  + MROWS * DMODEL;             // 4096 x 256
    float* xz   = hn   + MROWS * DMODEL;             // 4096 x 1024
    float* xc   = xz   + MROWS * 2 * DINNER;         // 4096 x 512
    float* dbc  = xc   + MROWS * DINNER;             // 4096 x 48
    float* dtv  = dbc  + MROWS * 48;                 // 4096 x 512
    float* ysb  = dtv  + MROWS * DINNER;             // 4096 x 512
    float* yco  = ysb  + MROWS * DINNER;             // 4096 x 512
    float* hmid = yco  + MROWS * DINNER;             // 4096 x 128

    const dim3 blk(256);
#define GEMM_BLOCKS(N, NT) ((MROWS / 16) * ((N) / (16 * (NT))) / 8)

    // embedding GEMM + bias + positional encoding, then LayerNorm (in-place)
    gemm_wmma_kernel<EPI_EMB, 4>
        <<<GEMM_BLOCKS(DMODEL, 4), blk, 0, stream>>>(
            x, emb_w, emb_b, nullptr, res, DMODEL, DIN);
    layernorm_kernel<<<MROWS, DMODEL, 0, stream>>>(res, ln_w, ln_b);

    for (int i = 0; i < NLAYER; ++i) {
        rmsnorm_kernel<<<MROWS, DMODEL, 0, stream>>>(res, norm_w + i * DMODEL, hn);

        gemm_wmma_kernel<EPI_STORE, 4>
            <<<GEMM_BLOCKS(2 * DINNER, 4), blk, 0, stream>>>(
                hn, in_proj + i * 2 * DINNER * DMODEL, nullptr, nullptr,
                xz, 2 * DINNER, DMODEL);

        conv_silu_kernel<<<MROWS * DINNER / 256, blk, 0, stream>>>(
            xz, conv_w + i * DINNER * DCONV, conv_b + i * DINNER, xc);

        gemm_wmma_kernel<EPI_STORE, 3>
            <<<GEMM_BLOCKS(48, 3), blk, 0, stream>>>(
                xc, x_proj + i * 48 * DINNER, nullptr, nullptr, dbc, 48, DINNER);

        dt_wmma_kernel<<<(MROWS / 16) * (DINNER / 64) / 8, blk, 0, stream>>>(
            dbc, dt_w + i * DINNER * DTRANK, dt_b + i * DINNER, dtv);

        scan_kernel<<<dim3(DINNER / 256, BATCH), blk, 0, stream>>>(
            dtv, xc, dbc, A_log + i * DINNER * NSTATE, ysb);

        combine_kernel<<<MROWS * DINNER / 256, blk, 0, stream>>>(
            ysb, xc, xz, Dp + i * DINNER, yco);

        gemm_wmma_kernel<EPI_ADD, 4>
            <<<GEMM_BLOCKS(DMODEL, 4), blk, 0, stream>>>(
                yco, out_proj + i * DMODEL * DINNER, nullptr, res, res,
                DMODEL, DINNER);
    }

    // final RMSNorm + 2-layer head
    rmsnorm_kernel<<<MROWS, DMODEL, 0, stream>>>(res, normf_w, hn);
    gemm_wmma_kernel<EPI_BIAS_RELU, 4>
        <<<GEMM_BLOCKS(128, 4), blk, 0, stream>>>(
            hn, h1_w, h1_b, nullptr, hmid, 128, DMODEL);
    gemm_wmma_kernel<EPI_BIAS, 4>
        <<<GEMM_BLOCKS(DIN, 4), blk, 0, stream>>>(
            hmid, h2_w, h2_b, nullptr, outp, DIN, 128);
}
